// reinforce_cons_loss_66709432042354
// MI455X (gfx1250) — matrix operand order repaired and checked by reading, not compiled, not run on hardware
//
#include <hip/hip_runtime.h>
#include <stdint.h>

#define B_    16
#define S_    10
#define CLS_  4
#define FG_   3
#define H_    192
#define W_    192
#define WPR   6                  // data words per bit-packed row
#define BST   8                  // padded word stride per row (1 guard each side)
#define BROWS (H_ + 16)          // 8 pad rows top + bottom
#define NBW   (BROWS * BST)      // 1664 words per padded bit buffer
#define NW    (H_ * WPR)         // 1152 interior words
#define NPIX  (H_ * W_)          // 36864
#define NTASK (B_ * S_ * FG_)    // 480
#define UST   208                // U buffer row stride (192 + 2*8 pad)

// padded bit-buffer word index for interior row r (0..191), word col wc (0..5)
#define BW(r, wc) (((r) + 8) * BST + 1 + (wc))

typedef __attribute__((ext_vector_type(16))) _Float16 v16h;
typedef __attribute__((ext_vector_type(8)))  _Float16 v8h;
typedef __attribute__((ext_vector_type(8)))  float    v8f;

struct Shm {
  uint32_t fgbits[NBW];
  uint32_t compA[NBW];
  uint32_t compB[NBW];
  alignas(16) _Float16 U[H_][UST];  // T@X intermediate, cols padded by 8/side
  float    red[256];
  uint32_t key;
  uint32_t flag[2];
};

// Banded conv3x3-ones as two WMMA passes: Y = T @ X @ T, T tridiagonal ones.
// Input X is a padded bit-packed mask `bits` (pad rows/words are zero, so no
// bounds checks are needed). Per-element result delivered to consume(row,col,y).
// Caller must __syncthreads() before calling; sh->U is clobbered.
template <class F>
__device__ inline void band_conv(Shm* sh, const uint32_t* bits,
                                 int lane, int wv, F&& consume) {
  const int hi = lane >> 4;      // lane half (0: lanes 0-15, 1: lanes 16-31)
  const int lm = lane & 15;

  // Constant banded A fragment (16x32): A[m,k] = 1 iff k in {m+7,m+8,m+9}.
  // 16-bit A layout: VGPR jj<4 -> K = hi*8 + 2jj + p ; jj>=4 -> K = 16 + hi*8 + 2(jj-4) + p
  v16h Ac;
#pragma unroll
  for (int e = 0; e < 16; ++e) {
    int jj = e >> 1, p = e & 1;
    int k = (jj < 4) ? (hi * 8 + 2 * jj + p) : (16 + hi * 8 + 2 * (jj - 4) + p);
    Ac[e] = (k >= lm + 7 && k <= lm + 9) ? (_Float16)1.0f : (_Float16)0.0f;
  }

  // ---- stage 1: U = T @ X (row conv). 144 tiles, 18 per wave. ----
  {
    int ti = (wv * 18) / 12, tj = (wv * 18) % 12;
    for (int q = 0; q < 18; ++q) {
      // B data fragment (32x16): B[k,n] = X[16*ti + k - 8][16*tj + n]
      // (padded row index = 16*ti + k, always in [0,207] -> branch-free)
      // 16-bit B layout (matches A): k = hi*8 + e (e<8), 16 + hi*8 + (e-8) (e>=8)
      const int col = 16 * tj + lm;
      const int wsel = col >> 5, bsel = col & 31;
      v16h Bd;
#pragma unroll
      for (int e = 0; e < 16; ++e) {
        int prow = 16 * ti + hi * 8 + e + ((e >= 8) ? 8 : 0); // padded row
        uint32_t bit = (bits[prow * BST + 1 + wsel] >> bsel) & 1u;
        Bd[e] = bit ? (_Float16)1.0f : (_Float16)0.0f;
      }
      v8f c = {};
      c = __builtin_amdgcn_wmma_f32_16x16x32_f16(false, Ac, false, Bd,
                                                 (short)0, c, false, false);
      // C layout: VGPR g -> M = hi*8 + g, N = lm. Values 0..3, exact in f16.
#pragma unroll
      for (int g = 0; g < 8; ++g) {
        int m = hi * 8 + g;
        sh->U[16 * ti + m][8 + 16 * tj + lm] = (_Float16)c[g];
      }
      if (++tj == 12) { tj = 0; ++ti; }
    }
  }
  __syncthreads();

  // Constant banded B fragment (32x16): B[k,n] = 1 iff k in {n+7,n+8,n+9}.
  v16h Bc;
#pragma unroll
  for (int e = 0; e < 16; ++e) {
    int k = hi * 8 + e + ((e >= 8) ? 8 : 0);
    Bc[e] = (k >= lm + 7 && k <= lm + 9) ? (_Float16)1.0f : (_Float16)0.0f;
  }

  // ---- stage 2: Y = U @ T (col conv), consume results. ----
  {
    int ti = (wv * 18) / 12, tj = (wv * 18) % 12;
    for (int q = 0; q < 18; ++q) {
      // A data fragment (16x32): A[m,k] = U[16*ti + m][16*tj + k] (padded cols).
      // Per lane this is two runs of 8 contiguous halves:
      //   e in [0,8):  k = hi*8 + e
      //   e in [8,16): k = 16 + hi*8 + (e-8)
      // Both 16B-aligned (row*UST*2 and col*2 are multiples of 16) -> 2x b128.
      const _Float16* abase = &sh->U[16 * ti + lm][16 * tj + hi * 8];
      v8h a0 = *reinterpret_cast<const v8h*>(abase);
      v8h a1 = *reinterpret_cast<const v8h*>(abase + 16);
      v16h Ad;
#pragma unroll
      for (int e = 0; e < 8; ++e) { Ad[e] = a0[e]; Ad[e + 8] = a1[e]; }

      v8f c = {};
      c = __builtin_amdgcn_wmma_f32_16x16x32_f16(false, Ad, false, Bc,
                                                 (short)0, c, false, false);
#pragma unroll
      for (int g = 0; g < 8; ++g) {
        int row = 16 * ti + hi * 8 + g;
        int col = 16 * tj + lm;
        consume(row, col, c[g]);
      }
      if (++tj == 12) { tj = 0; ++ti; }
    }
  }
}

__global__ __launch_bounds__(256) void conn_loss_kernel(
    const float* __restrict__ preds, const int* __restrict__ samples,
    float* __restrict__ contrib) {
  __shared__ Shm sh;
  const int tid  = threadIdx.x;
  const int lane = tid & 31;
  const int wv   = tid >> 5;

  const int task = blockIdx.x;
  const int b    = task / (S_ * FG_);
  const int rem  = task % (S_ * FG_);
  const int s    = rem / FG_;
  const int cls  = rem % FG_ + 1;

  const int*   splane = samples + (size_t)(b * S_ + s) * NPIX;
  const float* pplane = preds + (size_t)(b * CLS_ + cls) * NPIX;

  // Prefetch the (only) preds plane we will gather from; hidden behind the
  // flood-fill iterations.
  for (int o = tid * 64; o < NPIX; o += 256 * 64)
    __builtin_prefetch(pplane + o, 0, 0);

  // ---- init: zero all three padded bit buffers (pads must stay zero) ----
  if (tid == 0) { sh.key = 36863u; sh.flag[0] = 0u; sh.flag[1] = 0u; }
  for (int w = tid; w < NBW; w += 256) {
    sh.fgbits[w] = 0u;
    sh.compA[w]  = 0u;
    sh.compB[w]  = 0u;
  }
  for (int idx = tid; idx < H_ * 16; idx += 256) {  // zero U pad columns
    int r = idx >> 4, cc = idx & 15;
    sh.U[r][(cc < 8) ? cc : (192 + cc)] = (_Float16)0.0f;
  }
  __syncthreads();

  // ---- build bit-packed fg mask (ballot: 32 lanes -> 1 word) ----
  for (int w = wv; w < NW; w += 8) {
    int row = w / WPR, wc = w % WPR;
    int v = splane[row * W_ + wc * 32 + lane];
    uint32_t m = __builtin_amdgcn_ballot_w32(v == cls);
    if (lane == 0) sh.fgbits[BW(row, wc)] = m;
  }
  __syncthreads();

  // ---- conv1 = conv3x3(fg); seed = first argmax of conv*fg ----
  band_conv(&sh, sh.fgbits, lane, wv, [&](int row, int col, float y) {
    uint32_t bit = (sh.fgbits[BW(row, col >> 5)] >> (col & 31)) & 1u;
    if (bit) {
      int v = (int)(y + 0.5f);   // exact integer 1..9
      uint32_t key = ((uint32_t)v << 16) | (36863u - (uint32_t)(row * W_ + col));
      atomicMax(&sh.key, key);   // higher count wins; ties -> lowest index
    }
  });
  __syncthreads();

  if (tid == 0) {
    int idx = 36863 - (int)(sh.key & 0xFFFFu);
    int row = idx / W_, col = idx % W_;
    int w = BW(row, col >> 5);
    sh.compA[w] = sh.fgbits[w] & (1u << (col & 31));  // m0 = seed * fg
  }
  __syncthreads();

  // ---- flood fill: comp = dilate8(comp) & fg until fixpoint (branch-free) ----
  uint32_t* cur = sh.compA;
  uint32_t* nxt = sh.compB;
  int it = 0;
  for (;;) {
    uint32_t changed = 0;
    for (int w = tid; w < NW; w += 256) {
      int row = w / WPR, wc = w % WPR;
      int p = BW(row, wc);
      uint32_t d = 0;
#pragma unroll
      for (int dr = -1; dr <= 1; ++dr) {
        int q = p + dr * BST;                 // pad rows/words are zero
        uint32_t c0 = cur[q];
        uint32_t le = cur[q - 1];
        uint32_t ri = cur[q + 1];
        d |= c0 | (c0 << 1) | (c0 >> 1) | (le >> 31) | (ri << 31);
      }
      uint32_t nw = d & sh.fgbits[p];
      changed |= (nw ^ cur[p]);
      nxt[p] = nw;
    }
    if (changed) atomicOr(&sh.flag[it & 1], 1u);
    if (tid == 0) sh.flag[(it + 1) & 1] = 0u;  // pre-clear next slot
    __syncthreads();
    uint32_t any = sh.flag[it & 1];
    if (!any) break;                            // uniform exit
    uint32_t* tmp = cur; cur = nxt; nxt = tmp;
    ++it;
    __syncthreads();
  }
  const uint32_t* comp = nxt;  // converged component (== cur content)
  __syncthreads();

  // ---- contribution: sum lp*per_c ; per_c = conv(comp)/9 on comp, -1 on fg\comp
  float acc = 0.0f;
  for (int w = tid; w < NW; w += 256) {
    int row = w / WPR, wc = w % WPR;
    int p = BW(row, wc);
    uint32_t iso = sh.fgbits[p] & ~comp[p];
    int cbase = wc * 32;
    while (iso) {
      int bpos = __builtin_ctz(iso);
      acc -= logf(pplane[row * W_ + cbase + bpos] + 1e-8f);
      iso &= iso - 1u;
    }
  }
  __syncthreads();

  band_conv(&sh, comp, lane, wv, [&](int row, int col, float y) {
    uint32_t bit = (comp[BW(row, col >> 5)] >> (col & 31)) & 1u;
    if (bit) {
      float lp = logf(pplane[row * W_ + col] + 1e-8f);
      acc += lp * y * (1.0f / 9.0f);
    }
  });

  // ---- fixed-order workgroup reduction -> per-task slot (deterministic) ----
  sh.red[tid] = acc;
  __syncthreads();
  for (int off = 128; off > 0; off >>= 1) {
    if (tid < off) sh.red[tid] += sh.red[tid + off];
    __syncthreads();
  }
  if (tid == 0) contrib[task] = sh.red[0];
}

__global__ __launch_bounds__(512) void finalize_kernel(
    const float* __restrict__ contrib, float* __restrict__ out) {
  __shared__ float red[512];
  const int t = threadIdx.x;
  red[t] = (t < NTASK) ? contrib[t] : 0.0f;
  __syncthreads();
  for (int off = 256; off > 0; off >>= 1) {
    if (t < off) red[t] += red[t + off];
    __syncthreads();
  }
  if (t == 0) out[0] = -red[0] / (float)(B_ * S_ * H_ * W_);
}

extern "C" void kernel_launch(void* const* d_in, const int* in_sizes, int n_in,
                              void* d_out, int out_size, void* d_ws,
                              size_t ws_size, hipStream_t stream) {
  (void)in_sizes; (void)n_in; (void)out_size; (void)ws_size;
  const float* preds   = (const float*)d_in[0];
  const int*   samples = (const int*)d_in[1];
  float*       ws      = (float*)d_ws;     // 480 per-task partial sums

  conn_loss_kernel<<<NTASK, 256, 0, stream>>>(preds, samples, ws);
  finalize_kernel<<<1, 512, 0, stream>>>(ws, (float*)d_out);
}
